// Critic_69140383531303
// MI455X (gfx1250) — compile-verified, matrix operand order checked by reading, not run on hardware
//
#include <hip/hip_runtime.h>
#include <hip/hip_bf16.h>

typedef __attribute__((ext_vector_type(16))) _Float16 v16h;
typedef __attribute__((ext_vector_type(8)))  _Float16 v8h;
typedef __attribute__((ext_vector_type(8)))  float    v8f;
typedef __attribute__((ext_vector_type(4)))  unsigned int v4u;
typedef __attribute__((ext_vector_type(8)))  int v8i;
typedef __attribute__((ext_vector_type(4)))  int v4i;

#define NNODE 3
#define OBSD  30
#define ACTD  5
#define HDIM  64
#define TS    64             // samples per workgroup
#define RROWS (TS * NNODE)   // 192 node-rows per workgroup

#if __has_builtin(__builtin_amdgcn_tensor_load_to_lds)
#define HAVE_TDM 1
#endif

#ifdef HAVE_TDM
// 1D contiguous tile DMA: global -> LDS via Tensor Data Mover (ISA 8.3/8.4).
// D# group0: count=1 | lds_addr | global_addr[56:0] | type=2.
// D# group1: data_size, tensor_dim0 = tile_dim0 = nelem (nelem < 65536), dim1 = 1.
// 6-arg builtin form (clang-23 / therock-10.0): groups 2/3 + extended group zeroed (2D tensor).
__device__ __forceinline__ void tdm_load_1d(void* lds_dst, const void* gsrc,
                                            unsigned int nelem, unsigned int dsz_log2) {
    unsigned long long ga = (unsigned long long)(uintptr_t)gsrc;
    unsigned int lo = (unsigned int)(uintptr_t)lds_dst;   // flat LDS addr low 32b = LDS offset
    v4u g0;
    g0[0] = 1u;                                            // count=1 (valid user descriptor)
    g0[1] = lo;                                            // lds_addr
    g0[2] = (unsigned int)ga;                              // global_addr[31:0]
    g0[3] = (unsigned int)(ga >> 32) | 0x80000000u;        // global_addr[56:32] | type=2
    v8i g1;
    g1[0] = (int)(dsz_log2 << 16);                         // data_size; no multicast/iterate/pad
    g1[1] = (int)(nelem << 16);                            // tensor_dim0[15:0]
    g1[2] = (int)((nelem >> 16) | (1u << 16));             // tensor_dim0[31:16], tensor_dim1=1
    g1[3] = (int)(nelem << 16);                            // tile_dim0 = nelem
    g1[4] = 0;                                             // tile_dim1 = tile_dim2 = 0
    g1[5] = (int)nelem;                                    // tensor_dim0_stride[31:0]
    g1[6] = 0; g1[7] = 0;
    v4i z4 = {0, 0, 0, 0};
    v8i z8 = {0, 0, 0, 0, 0, 0, 0, 0};
    __builtin_amdgcn_tensor_load_to_lds(g0, g1, z4, z4, z8, 0);
}
#endif

// ---- WMMA fragment helpers (CDNA5 ISA 7.12.2 layouts) ----
__device__ __forceinline__ v16h frag_ld(const _Float16* base, int ld, int row0, int k0, int lane) {
    int r  = row0 + (lane & 15);
    int kh = (lane >> 4) << 3;
    const _Float16* p = base + r * ld + k0 + kh;
    v8h lo = *(const v8h*)p;          // 16B contiguous -> ds_load_b128
    v8h hi = *(const v8h*)(p + 16);
    v16h f;
#pragma unroll
    for (int i = 0; i < 8; ++i) { f[i] = lo[i]; f[8 + i] = hi[i]; }
    return f;
}
__device__ __forceinline__ void frag_st_f32(float* base, int ld, int row0, int col0, int lane, v8f acc) {
    int m0 = row0 + ((lane >> 4) << 3);
    int n  = col0 + (lane & 15);
#pragma unroll
    for (int v = 0; v < 8; ++v) base[(m0 + v) * ld + n] = acc[v];
}
__device__ __forceinline__ void frag_st_f16(_Float16* base, int ld, int row0, int col0, int lane, v8f acc) {
    int m0 = row0 + ((lane >> 4) << 3);
    int n  = col0 + (lane & 15);
#pragma unroll
    for (int v = 0; v < 8; ++v) base[(m0 + v) * ld + n] = (_Float16)acc[v];
}

// =====================================================================
// Kernel 1: GAT layer 1 + GAT layer 2  ->  H2 [B, 192] fp16 (workspace)
// =====================================================================
__global__ __launch_bounds__(256) void gat_kernel(
    const float* __restrict__ obs, const float* __restrict__ act,
    const int* __restrict__ adj,
    const float* __restrict__ W1, const float* __restrict__ a1s,
    const float* __restrict__ a1d, const float* __restrict__ b1,
    const float* __restrict__ W2, const float* __restrict__ a2s,
    const float* __restrict__ a2d, const float* __restrict__ b2,
    _Float16* __restrict__ h2out)
{
    __shared__ __align__(16) _Float16 Xh[RROWS * 64];   // 24 KB, K padded 35->64
    __shared__ __align__(16) _Float16 W1t[64 * 64];     //  8 KB  [n][k]
    __shared__ __align__(16) _Float16 W2t[64 * 64];     //  8 KB  [n][k]
    __shared__ __align__(16) float    Z[RROWS * 64];    // 48 KB fp32; aliased as raw input staging
    __shared__ __align__(16) _Float16 Hh[RROWS * 64];   // 24 KB
    __shared__ __align__(16) int      adjs[TS * 9];     // 2.25 KB staged adjacency
    __shared__ float ssrc[RROWS], sdst[RROWS];
    __shared__ float alpha[TS * 9];
    __shared__ float va1s[64], va1d[64], vb1[64], va2s[64], va2d[64], vb2[64];

    const int tid  = threadIdx.x;
    const int lane = tid & 31;
    const int wave = tid >> 5;
    const int s0   = blockIdx.x * TS;

    // ---- async-stage raw obs/act/adj tiles into LDS ----
#ifdef HAVE_TDM
    if (wave == 0) {   // TDM ignores EXEC; issue once per workgroup (wave 0 only)
        tdm_load_1d(Z,            obs + (size_t)s0 * (NNODE * OBSD), TS * NNODE * OBSD, 2);
        tdm_load_1d(Z + 1440 * 4, act + (size_t)s0 * (NNODE * ACTD), TS * NNODE * ACTD, 2);
        tdm_load_1d(adjs,         adj + (size_t)s0 * 9,              TS * 9,            2);
    }
#else
    {
        float4* zs = (float4*)Z;
        const float4* osrc = (const float4*)(obs + (size_t)s0 * (NNODE * OBSD)); // 5760 f32
        const float4* asrc = (const float4*)(act + (size_t)s0 * (NNODE * ACTD)); //  960 f32
        for (int i = tid; i < 1440; i += 256) zs[i] = osrc[i];
        for (int i = tid; i < 240;  i += 256) zs[1440 + i] = asrc[i];
        const uint4* jsrc = (const uint4*)(adj + (size_t)s0 * 9);                //  576 i32
        uint4* jdst = (uint4*)adjs;
        for (int i = tid; i < 144; i += 256) jdst[i] = jsrc[i];
    }
#endif
    // ---- stage weights (transposed, fp16) with vectorized b128 loads ----
    for (int idx = tid; idx < 64 * 16; idx += 256) {      // 64 k-rows x 16 float4
        int k = idx >> 4, n4 = (idx & 15) << 2;
        float4 v1 = (k < OBSD + ACTD) ? *(const float4*)(W1 + k * 64 + n4)
                                      : make_float4(0.f, 0.f, 0.f, 0.f);
        float4 v2 = *(const float4*)(W2 + k * 64 + n4);
        W1t[(n4 + 0) * 64 + k] = (_Float16)v1.x; W1t[(n4 + 1) * 64 + k] = (_Float16)v1.y;
        W1t[(n4 + 2) * 64 + k] = (_Float16)v1.z; W1t[(n4 + 3) * 64 + k] = (_Float16)v1.w;
        W2t[(n4 + 0) * 64 + k] = (_Float16)v2.x; W2t[(n4 + 1) * 64 + k] = (_Float16)v2.y;
        W2t[(n4 + 2) * 64 + k] = (_Float16)v2.z; W2t[(n4 + 3) * 64 + k] = (_Float16)v2.w;
    }
    if (tid < 64) {
        va1s[tid] = a1s[tid]; va1d[tid] = a1d[tid]; vb1[tid] = b1[tid];
        va2s[tid] = a2s[tid]; va2d[tid] = a2d[tid]; vb2[tid] = b2[tid];
    }
    // ---- prefetch next block's input tiles toward L2 while we compute ----
    if (blockIdx.x + 1 < gridDim.x) {
        const char* nobs = (const char*)(obs + (size_t)(s0 + TS) * (NNODE * OBSD));
        __builtin_prefetch(nobs + tid * 128, 0, 1);     // 32 KB span covers 23 KB tile
        if (tid < 32) {
            const char* nact = (const char*)(act + (size_t)(s0 + TS) * (NNODE * ACTD));
            __builtin_prefetch(nact + tid * 128, 0, 1); // 4 KB span
            const char* nadj = (const char*)(adj + (size_t)(s0 + TS) * 9);
            __builtin_prefetch(nadj + tid * 128, 0, 1); // 4 KB span covers 2.3 KB
        }
    }
#ifdef HAVE_TDM
    if (wave == 0) __builtin_amdgcn_s_wait_tensorcnt(0);
#endif
    __syncthreads();
    // ---- pack X tile: concat(obs, action) padded to 64 cols, fp16 ----
    for (int i = tid; i < RROWS * 64; i += 256) {
        int r = i >> 6, c = i & 63;
        int sl = r / 3, node = r - sl * 3;
        float v = 0.f;
        if (c < OBSD)             v = Z[sl * (NNODE * OBSD) + node * OBSD + c];
        else if (c < OBSD + ACTD) v = Z[1440 * 4 + sl * (NNODE * ACTD) + node * ACTD + (c - OBSD)];
        Xh[i] = (_Float16)v;
    }
    __syncthreads();

    // ---- GEMM1: Z = X @ W1   (192x64 @ 64x64, K padded) ----
    for (int t = wave; t < 12 * 4; t += 8) {
        int mt = t >> 2, nt = t & 3;
        v8f acc = {};
#pragma unroll
        for (int ks = 0; ks < 2; ++ks) {
            v16h a = frag_ld(Xh,  64, mt * 16, ks * 32, lane);
            v16h b = frag_ld(W1t, 64, nt * 16, ks * 32, lane);
            acc = __builtin_amdgcn_wmma_f32_16x16x32_f16(false, a, false, b, (short)0, acc, false, false);
        }
        frag_st_f32(Z, 64, mt * 16, nt * 16, lane, acc);
    }
    __syncthreads();

    // ---- attention 1: scores ----
    if (tid < RROWS) {
        float as = 0.f, ad = 0.f;
        for (int d = 0; d < 64; ++d) { float z = Z[tid * 64 + d]; as += z * va1s[d]; ad += z * va1d[d]; }
        ssrc[tid] = as; sdst[tid] = ad;
    }
    __syncthreads();
    // ---- attention 1: multiplicity-weighted softmax ----
    if (tid < TS) {
        float cnt[9], e[9];
        for (int i = 0; i < 3; ++i)
            for (int j = 0; j < 3; ++j) {
                cnt[i * 3 + j] = (float)(adjs[tid * 9 + j * 3 + i] + (i == j ? 1 : 0));
                float ee = ssrc[tid * 3 + j] + sdst[tid * 3 + i];
                e[i * 3 + j] = ee > 0.f ? ee : 0.2f * ee;
            }
        for (int i = 0; i < 3; ++i) {
            float m = -1e30f;
            for (int j = 0; j < 3; ++j) if (cnt[i * 3 + j] > 0.f && e[i * 3 + j] > m) m = e[i * 3 + j];
            float p[3], sm = 0.f;
            for (int j = 0; j < 3; ++j) { p[j] = cnt[i * 3 + j] * __expf(e[i * 3 + j] - m); sm += p[j]; }
            float inv = 1.f / sm;
            for (int j = 0; j < 3; ++j) alpha[tid * 9 + i * 3 + j] = p[j] * inv;
        }
    }
    __syncthreads();
    // ---- aggregate + bias + ELU -> Hh fp16 ----
    for (int i = tid; i < RROWS * 64; i += 256) {
        int r = i >> 6, d = i & 63;
        int sl = r / 3, node = r - sl * 3;
        float acc = vb1[d];
        for (int j = 0; j < 3; ++j) acc += alpha[sl * 9 + node * 3 + j] * Z[(sl * 3 + j) * 64 + d];
        acc = acc > 0.f ? acc : (__expf(acc) - 1.f);
        Hh[i] = (_Float16)acc;
    }
    __syncthreads();

    // ---- GEMM2: Z = H1 @ W2 ----
    for (int t = wave; t < 12 * 4; t += 8) {
        int mt = t >> 2, nt = t & 3;
        v8f acc = {};
#pragma unroll
        for (int ks = 0; ks < 2; ++ks) {
            v16h a = frag_ld(Hh,  64, mt * 16, ks * 32, lane);
            v16h b = frag_ld(W2t, 64, nt * 16, ks * 32, lane);
            acc = __builtin_amdgcn_wmma_f32_16x16x32_f16(false, a, false, b, (short)0, acc, false, false);
        }
        frag_st_f32(Z, 64, mt * 16, nt * 16, lane, acc);
    }
    __syncthreads();

    // ---- attention 2 ----
    if (tid < RROWS) {
        float as = 0.f, ad = 0.f;
        for (int d = 0; d < 64; ++d) { float z = Z[tid * 64 + d]; as += z * va2s[d]; ad += z * va2d[d]; }
        ssrc[tid] = as; sdst[tid] = ad;
    }
    __syncthreads();
    if (tid < TS) {
        float cnt[9], e[9];
        for (int i = 0; i < 3; ++i)
            for (int j = 0; j < 3; ++j) {
                cnt[i * 3 + j] = (float)(adjs[tid * 9 + j * 3 + i] + (i == j ? 1 : 0));
                float ee = ssrc[tid * 3 + j] + sdst[tid * 3 + i];
                e[i * 3 + j] = ee > 0.f ? ee : 0.2f * ee;
            }
        for (int i = 0; i < 3; ++i) {
            float m = -1e30f;
            for (int j = 0; j < 3; ++j) if (cnt[i * 3 + j] > 0.f && e[i * 3 + j] > m) m = e[i * 3 + j];
            float p[3], sm = 0.f;
            for (int j = 0; j < 3; ++j) { p[j] = cnt[i * 3 + j] * __expf(e[i * 3 + j] - m); sm += p[j]; }
            float inv = 1.f / sm;
            for (int j = 0; j < 3; ++j) alpha[tid * 9 + i * 3 + j] = p[j] * inv;
        }
    }
    __syncthreads();
    // ---- aggregate 2 + bias (no activation) -> global H2 fp16 ----
    for (int i = tid; i < RROWS * 64; i += 256) {
        int r = i >> 6, d = i & 63;
        int sl = r / 3, node = r - sl * 3;
        float acc = vb2[d];
        for (int j = 0; j < 3; ++j) acc += alpha[sl * 9 + node * 3 + j] * Z[(sl * 3 + j) * 64 + d];
        h2out[(size_t)(s0 + sl) * 192 + node * 64 + d] = (_Float16)acc;
    }
}

// =====================================================================
// Kernel 2: MLP head  H2[B,192] -> out[B]
// =====================================================================
__global__ __launch_bounds__(256) void mlp_kernel(
    const _Float16* __restrict__ h2,
    const float* __restrict__ Wo1, const float* __restrict__ bo1,
    const float* __restrict__ g1,  const float* __restrict__ be1,
    const float* __restrict__ Wo2, const float* __restrict__ bo2,
    const float* __restrict__ g2,  const float* __restrict__ be2,
    const float* __restrict__ Wo3, const float* __restrict__ bo3,
    float* __restrict__ out)
{
    __shared__ __align__(16) _Float16 Wo1t[256 * 192];  // 96 KB  [n][k]
    __shared__ __align__(16) _Float16 Wo2t[128 * 256];  // 64 KB  [n][k]
    __shared__ __align__(16) _Float16 Ah[TS * 192];     // 24 KB input tile
    __shared__ __align__(16) _Float16 Y1h[TS * 256];    // 32 KB (pre-LN, then post-LN in place)
    __shared__ __align__(16) _Float16 Y2h[TS * 128];    // 16 KB
    __shared__ float vbo1[256], vg1[256], vbe1[256];
    __shared__ float vbo2[128], vg2[128], vbe2[128];
    __shared__ float vWo3[128];
    __shared__ float redS[4 * TS], redQ[4 * TS];        // LN partial sums
    __shared__ float rmu[TS], rrs[TS];                  // per-row mean / rstd

    const int tid  = threadIdx.x;
    const int lane = tid & 31;
    const int wave = tid >> 5;
    const int s0   = blockIdx.x * TS;

    // ---- async-stage input tile via TDM while other waves stage weights ----
#ifdef HAVE_TDM
    if (wave == 0) {
        tdm_load_1d(Ah, h2 + (size_t)s0 * 192, TS * 192, 1);   // 12288 x fp16
    }
#else
    {
        const uint4* src = (const uint4*)(h2 + (size_t)s0 * 192);
        uint4* dst = (uint4*)Ah;
        for (int i = tid; i < TS * 192 / 8; i += 256) dst[i] = src[i];
    }
#endif
    // ---- stage weights (transposed fp16), vectorized b128 loads ----
    for (int idx = tid; idx < 192 * 64; idx += 256) {   // Wo1: 192 k-rows x 64 float4
        int k = idx >> 6, n4 = (idx & 63) << 2;
        float4 v = *(const float4*)(Wo1 + k * 256 + n4);
        Wo1t[(n4 + 0) * 192 + k] = (_Float16)v.x; Wo1t[(n4 + 1) * 192 + k] = (_Float16)v.y;
        Wo1t[(n4 + 2) * 192 + k] = (_Float16)v.z; Wo1t[(n4 + 3) * 192 + k] = (_Float16)v.w;
    }
    for (int idx = tid; idx < 256 * 32; idx += 256) {   // Wo2: 256 k-rows x 32 float4
        int k = idx >> 5, n4 = (idx & 31) << 2;
        float4 v = *(const float4*)(Wo2 + k * 128 + n4);
        Wo2t[(n4 + 0) * 256 + k] = (_Float16)v.x; Wo2t[(n4 + 1) * 256 + k] = (_Float16)v.y;
        Wo2t[(n4 + 2) * 256 + k] = (_Float16)v.z; Wo2t[(n4 + 3) * 256 + k] = (_Float16)v.w;
    }
    if (tid < 256) { vbo1[tid] = bo1[tid]; vg1[tid] = g1[tid]; vbe1[tid] = be1[tid]; }
    if (tid < 128) { vbo2[tid] = bo2[tid]; vg2[tid] = g2[tid]; vbe2[tid] = be2[tid]; vWo3[tid] = Wo3[tid]; }
    // ---- prefetch next block's H2 tile ----
    if (blockIdx.x + 1 < gridDim.x) {
        const char* nxt = (const char*)(h2 + (size_t)(s0 + TS) * 192);
        __builtin_prefetch(nxt + tid * 128, 0, 1);       // 32 KB span covers 24 KB tile
    }
#ifdef HAVE_TDM
    if (wave == 0) __builtin_amdgcn_s_wait_tensorcnt(0);
#endif
    __syncthreads();

    // ---- GEMM1: Y1 = A(64x192) @ Wo1(192x256) ----
    for (int t = wave; t < 4 * 16; t += 8) {
        int mt = t >> 4, nt = t & 15;
        v8f acc = {};
#pragma unroll
        for (int ks = 0; ks < 6; ++ks) {
            v16h a = frag_ld(Ah,   192, mt * 16, ks * 32, lane);
            v16h b = frag_ld(Wo1t, 192, nt * 16, ks * 32, lane);
            acc = __builtin_amdgcn_wmma_f32_16x16x32_f16(false, a, false, b, (short)0, acc, false, false);
        }
        frag_st_f16(Y1h, 256, mt * 16, nt * 16, lane, acc);
    }
    __syncthreads();

    // ---- LayerNorm 1 (+bias) + leaky-ReLU(0.01), parallel over 4 col-parts ----
    {
        int r = tid & 63, p = tid >> 6;                 // 4 parts x 64 rows
        float s = 0.f, q = 0.f;
        for (int c = p * 64; c < p * 64 + 64; ++c) {
            float x = (float)Y1h[r * 256 + c] + vbo1[c];
            s += x; q += x * x;
        }
        redS[p * TS + r] = s; redQ[p * TS + r] = q;
    }
    __syncthreads();
    if (tid < TS) {
        float s = redS[tid] + redS[TS + tid] + redS[2 * TS + tid] + redS[3 * TS + tid];
        float q = redQ[tid] + redQ[TS + tid] + redQ[2 * TS + tid] + redQ[3 * TS + tid];
        float mu = s * (1.f / 256.f);
        float var = q * (1.f / 256.f) - mu * mu;
        rmu[tid] = mu; rrs[tid] = rsqrtf(var + 1e-5f);
    }
    __syncthreads();
    for (int i = tid; i < TS * 256; i += 256) {
        int r = i >> 8, c = i & 255;
        float x = ((float)Y1h[i] + vbo1[c] - rmu[r]) * rrs[r] * vg1[c] + vbe1[c];
        Y1h[i] = (_Float16)(x > 0.f ? x : 0.01f * x);
    }
    __syncthreads();

    // ---- GEMM2: Y2 = Y1(64x256) @ Wo2(256x128) ----
    for (int t = wave; t < 4 * 8; t += 8) {
        int mt = t >> 3, nt = t & 7;
        v8f acc = {};
#pragma unroll
        for (int ks = 0; ks < 8; ++ks) {
            v16h a = frag_ld(Y1h,  256, mt * 16, ks * 32, lane);
            v16h b = frag_ld(Wo2t, 256, nt * 16, ks * 32, lane);
            acc = __builtin_amdgcn_wmma_f32_16x16x32_f16(false, a, false, b, (short)0, acc, false, false);
        }
        frag_st_f16(Y2h, 128, mt * 16, nt * 16, lane, acc);
    }
    __syncthreads();

    // ---- LayerNorm 2 (+bias) + leaky-ReLU, parallel over 2 col-parts ----
    if (tid < 2 * TS) {
        int r = tid & 63, p = tid >> 6;                 // 2 parts x 64 rows
        float s = 0.f, q = 0.f;
        for (int c = p * 64; c < p * 64 + 64; ++c) {
            float x = (float)Y2h[r * 128 + c] + vbo2[c];
            s += x; q += x * x;
        }
        redS[p * TS + r] = s; redQ[p * TS + r] = q;
    }
    __syncthreads();
    if (tid < TS) {
        float s = redS[tid] + redS[TS + tid];
        float q = redQ[tid] + redQ[TS + tid];
        float mu = s * (1.f / 128.f);
        float var = q * (1.f / 128.f) - mu * mu;
        rmu[tid] = mu; rrs[tid] = rsqrtf(var + 1e-5f);
    }
    __syncthreads();
    for (int i = tid; i < TS * 128; i += 256) {
        int r = i >> 7, c = i & 127;
        float x = ((float)Y2h[i] + vbo2[c] - rmu[r]) * rrs[r] * vg2[c] + vbe2[c];
        Y2h[i] = (_Float16)(x > 0.f ? x : 0.01f * x);
    }
    __syncthreads();

    // ---- final projection 128 -> 1 ----
    if (tid < TS) {
        float acc = bo3[0];
        for (int c = 0; c < 128; ++c) acc += (float)Y2h[tid * 128 + c] * vWo3[c];
        out[s0 + tid] = acc;
    }
}

extern "C" void kernel_launch(void* const* d_in, const int* in_sizes, int n_in,
                              void* d_out, int out_size, void* d_ws, size_t ws_size,
                              hipStream_t stream) {
    const float* obs  = (const float*)d_in[0];
    const float* act  = (const float*)d_in[1];
    const int*   adj  = (const int*)d_in[2];
    const float* W1   = (const float*)d_in[3];
    const float* a1s  = (const float*)d_in[4];
    const float* a1d  = (const float*)d_in[5];
    const float* b1   = (const float*)d_in[6];
    const float* W2   = (const float*)d_in[7];
    const float* a2s  = (const float*)d_in[8];
    const float* a2d  = (const float*)d_in[9];
    const float* b2   = (const float*)d_in[10];
    const float* Wo1  = (const float*)d_in[11];
    const float* bo1  = (const float*)d_in[12];
    const float* g1   = (const float*)d_in[13];
    const float* be1  = (const float*)d_in[14];
    const float* Wo2  = (const float*)d_in[15];
    const float* bo2  = (const float*)d_in[16];
    const float* g2   = (const float*)d_in[17];
    const float* be2  = (const float*)d_in[18];
    const float* Wo3  = (const float*)d_in[19];
    const float* bo3  = (const float*)d_in[20];
    float* out = (float*)d_out;

    const int Bsz = in_sizes[0] / (NNODE * OBSD);   // 262144
    _Float16* h2 = (_Float16*)d_ws;                 // [B, 192] fp16
    const int nblk = Bsz / TS;

    gat_kernel<<<nblk, 256, 0, stream>>>(obs, act, adj, W1, a1s, a1d, b1,
                                         W2, a2s, a2d, b2, h2);
    mlp_kernel<<<nblk, 256, 0, stream>>>(h2, Wo1, bo1, g1, be1,
                                         Wo2, bo2, g2, be2, Wo3, bo3, out);
}
